// QuantizedLinear_37452114821241
// MI455X (gfx1250) — matrix operand order repaired and checked by reading, not compiled, not run on hardware
//
#include <hip/hip_runtime.h>

// QuantizedLinear on gfx1250 (MI455X):
//   y[m, o] = sum_k x[m,k] * (float)qw[o,k] * scale[o] + bias[o]
// M = B*S = 512, K = IN = 4096, N = OUT = 11008.
//
// Roofline: weights = 180 MB int32 (dominant, read-once target), x = 8.4 MB,
// y = 22.5 MB -> ~9 us floor @ 23.3 TB/s. f16 WMMA keeps compute under that
// (int8-valued weights convert to f16 exactly; x only loses fp32->fp16 rounding).
//
//  - v_wmma_f32_16x16x32_f16, f32 accumulators; K-step = 32.
//  - Per wave: 4 M-tiles x 2 N-tiles (64x32 output). Block = 8 waves = 256 N.
//  - grid.x = M-blocks (fast dispatch axis) so the 8 blocks sharing a weight
//    stripe are dispatch-adjacent -> weight stripe read once from HBM, 7x from L2.
//  - Non-temporal stores for Y (write-once) so output doesn't evict the
//    weight working set from the 192 MB L2.

#define IN_DIM   4096
#define OUT_DIM  11008
#define M_ROWS   512   // B*S = 8*64

typedef __attribute__((ext_vector_type(16))) _Float16 v16h;
typedef __attribute__((ext_vector_type(8)))  float    v8f;
typedef __attribute__((ext_vector_type(4)))  float    v4f;
typedef __attribute__((ext_vector_type(4)))  int      v4i;

__global__ __launch_bounds__(256)
void QuantizedLinear_37452114821241_kernel(const float* __restrict__ X,
                                           const int*   __restrict__ W,
                                           const float* __restrict__ scale,
                                           const float* __restrict__ bias,
                                           float*       __restrict__ Y)
{
    const int tid   = threadIdx.x;
    const int wave  = tid >> 5;        // 0..7
    const int lane  = tid & 31;        // wave32 lane
    const int laneM = lane & 15;       // 0..15
    const int lhalf = lane >> 4;       // 0 or 1

    // x = M (fast dispatch axis -> L2 reuse of weight stripes), y = N.
    const int m0 = blockIdx.x * 64;                 // 4 M-tiles of 16
    const int n0 = blockIdx.y * 256 + wave * 32;    // 2 N-tiles of 16 per wave

    // Row pointers for the A fragments (M = laneM within each 16-row tile).
    const float* xrow[4];
#pragma unroll
    for (int mt = 0; mt < 4; ++mt)
        xrow[mt] = X + (size_t)(m0 + mt * 16 + laneM) * IN_DIM;

    // Row pointers for the B fragments: B element (k,n) = W[n, k];
    // lane holds N = laneM, K = lhalf*16 + h (16 consecutive int32 per lane).
    // Lanes l and l+16 read the two halves of the same 128B line -> full
    // cacheline utilization on the weight stream.
    const int* wrow[2];
#pragma unroll
    for (int nt = 0; nt < 2; ++nt)
        wrow[nt] = W + (size_t)(n0 + nt * 16 + laneM) * IN_DIM + lhalf * 16;

    v8f acc[4][2] = {};   // f32 accumulators (8 VGPRs each)

    for (int k0 = 0; k0 < IN_DIM; k0 += 32) {
        // ---- B fragments: 32x16 int8-valued weights -> f16 (exact) ----
        v16h bfrag[2];
#pragma unroll
        for (int nt = 0; nt < 2; ++nt) {
            const int* p = wrow[nt] + k0;
            v4i w0 = *(const v4i*)(p + 0);
            v4i w1 = *(const v4i*)(p + 4);
            v4i w2 = *(const v4i*)(p + 8);
            v4i w3 = *(const v4i*)(p + 12);
#pragma unroll
            for (int j = 0; j < 4; ++j) {
                bfrag[nt][j +  0] = (_Float16)w0[j];
                bfrag[nt][j +  4] = (_Float16)w1[j];
                bfrag[nt][j +  8] = (_Float16)w2[j];
                bfrag[nt][j + 12] = (_Float16)w3[j];
            }
            // keep the dominant (weight) stream ahead of the loads
            __builtin_prefetch(p + 128, 0, 1);
        }

        // ---- A fragments: 16x32 fp32 activations -> f16 ----
        // lane holds M = laneM; halves 0..15 <- k0 + lhalf*16 .. +15
        v16h afrag[4];
#pragma unroll
        for (int mt = 0; mt < 4; ++mt) {
            const float* p = xrow[mt] + k0 + lhalf * 16;
            v4f x0 = *(const v4f*)(p + 0);
            v4f x1 = *(const v4f*)(p + 4);
            v4f x2 = *(const v4f*)(p + 8);
            v4f x3 = *(const v4f*)(p + 12);
#pragma unroll
            for (int j = 0; j < 4; ++j) {
                afrag[mt][j +  0] = (_Float16)x0[j];
                afrag[mt][j +  4] = (_Float16)x1[j];
                afrag[mt][j +  8] = (_Float16)x2[j];
                afrag[mt][j + 12] = (_Float16)x3[j];
            }
        }

        // ---- 8 WMMAs: D = A x B + C ----
#pragma unroll
        for (int mt = 0; mt < 4; ++mt)
#pragma unroll
            for (int nt = 0; nt < 2; ++nt)
                acc[mt][nt] = __builtin_amdgcn_wmma_f32_16x16x32_f16(
                    /*neg_a=*/false, afrag[mt],
                    /*neg_b=*/false, bfrag[nt],
                    /*c_mod=*/(short)0, acc[mt][nt],
                    /*reuse_a=*/false, /*reuse_b=*/false);
    }

    // ---- Epilogue: per-column scale + bias, non-temporal f32 stores ----
    // C layout: VGPR v of lane l holds (row = v + (l/16)*8, col = l%16).
#pragma unroll
    for (int nt = 0; nt < 2; ++nt) {
        const int col = n0 + nt * 16 + laneM;
        const float s = scale[col];
        const float b = bias[col];
#pragma unroll
        for (int mt = 0; mt < 4; ++mt) {
            const int rbase = m0 + mt * 16 + lhalf * 8;
#pragma unroll
            for (int v = 0; v < 8; ++v) {
                float r = acc[mt][nt][v] * s + b;
                // write-once output: NT store, don't evict weights from L2
                __builtin_nontemporal_store(r, &Y[(size_t)(rbase + v) * OUT_DIM + col]);
            }
        }
    }
}

extern "C" void kernel_launch(void* const* d_in, const int* in_sizes, int n_in,
                              void* d_out, int out_size, void* d_ws, size_t ws_size,
                              hipStream_t stream) {
    const float* x  = (const float*)d_in[0];   // [8, 64, 4096] fp32
    const int*   qw = (const int*)  d_in[1];   // [11008, 4096] int32 (int8-valued)
    const float* sc = (const float*)d_in[2];   // [11008, 1] fp32
    const float* bi = (const float*)d_in[3];   // [11008] fp32
    float*       y  = (float*)d_out;           // [8, 64, 11008] fp32

    dim3 grid(M_ROWS / 64, OUT_DIM / 256, 1);  // (8 M-blocks fast) x (43 N-stripes)
    dim3 block(256, 1, 1);                      // 8 waves (wave32)
    hipLaunchKernelGGL(QuantizedLinear_37452114821241_kernel,
                       grid, block, 0, stream, x, qw, sc, bi, y);
}
